// AdditiveAttention_42056319762945
// MI455X (gfx1250) — matrix operand order repaired and checked by reading, not compiled, not run on hardware
//
#include <hip/hip_runtime.h>

// ---------- types for WMMA fragments (wave32, gfx1250) ----------
typedef __attribute__((ext_vector_type(16))) __bf16       v16bf;
typedef __attribute__((ext_vector_type(8)))  float        v8f;
typedef __attribute__((ext_vector_type(4)))  unsigned int v4u;

union Frag16 { v16bf v; v4u q[2]; };

// Problem constants (from reference): B=8, Q=256, K=512, D=256, H=128, V=256
#define BB 8
#define QQ 256
#define KK 512
#define DD 256
#define HH 128
#define VV 256
#define QTILE 8   // q-rows per block in the fused score/softmax kernel

__device__ __forceinline__ __bf16 f2bf(float f) {
  union { float f; unsigned u; } x; x.f = f;
  unsigned r = x.u + 0x7FFFu + ((x.u >> 16) & 1u);   // round-to-nearest-even
  union { unsigned short s; __bf16 b; } y;
  y.s = (unsigned short)(r >> 16);
  return y.b;
}

// tanh via the fast exp pipe, overflow-safe: tanh(x)=sign(x)*(1-e^{-2|x|})/(1+e^{-2|x|})
__device__ __forceinline__ float fast_tanh(float x) {
  float ax = fabsf(x);
  float t  = __expf(-2.0f * ax);
  float r  = (1.0f - t) / (1.0f + t);
  return copysignf(r, x);
}

// ---------- elementwise fp32 -> bf16 ----------
__global__ void __launch_bounds__(256) cvt_f32_bf16(const float* __restrict__ s,
                                                    __bf16* __restrict__ d, int n) {
  int i = blockIdx.x * blockDim.x + threadIdx.x;
  if (i < n) d[i] = f2bf(s[i]);
}

// value [B,K,V] fp32 -> valueT [B,V,K] bf16 (so the AV GEMM reads B^T rows K-contiguous)
__global__ void __launch_bounds__(256) transpose_cvt_value(const float* __restrict__ v,
                                                           __bf16* __restrict__ vt) {
  int i = blockIdx.x * blockDim.x + threadIdx.x;   // over B*K*V = 1,048,576
  if (i >= BB * KK * VV) return;
  int c = i & (VV - 1);
  int k = (i >> 8) & (KK - 1);
  int b = i >> 17;
  vt[((long long)((b << 8) + c)) * KK + k] = f2bf(v[i]);
}

// ---------- generic bf16 WMMA GEMM: C[m][n] = sum_k A[m][k] * Bt[n][k] ----------
// A: row-major [M,Kd] bf16; Bt: row-major [N,Kd] bf16 (i.e. B transposed); C: fp32 [M,N].
// One wave computes one 16x16 tile, K stepped by 32 via v_wmma_f32_16x16x32_bf16.
__global__ void __launch_bounds__(256)
gemm_bf16_wmma(const __bf16* __restrict__ A, long long strideA,
               const __bf16* __restrict__ Bt, long long strideB,
               float* __restrict__ C, long long strideC,
               int M, int N, int Kd) {
  const int batch = blockIdx.y;
  A  += (long long)batch * strideA;
  Bt += (long long)batch * strideB;
  C  += (long long)batch * strideC;

  const int lane = threadIdx.x & 31;
  const int wave = threadIdx.x >> 5;
  const int tilesN = N >> 4;
  const int tile = blockIdx.x * 8 + wave;
  if (tile >= (M >> 4) * tilesN) return;     // uniform per wave: EXEC stays all-1s for WMMA
  const int tm = (tile / tilesN) << 4;
  const int tn = (tile % tilesN) << 4;

  const int half = lane >> 4;                // lane half selects K sub-block (ISA 16-bit layouts)
  const int lm   = lane & 15;                // row (A) / col (B) index

  const __bf16* arow = A  + (long long)(tm + lm) * Kd;
  const __bf16* brow = Bt + (long long)(tn + lm) * Kd;

  v8f acc = {};
  for (int k0 = 0; k0 < Kd; k0 += 32) {
    Frag16 a, b;
    // A 16x32 bf16 layout: lanes<16 hold K {0..7,16..23}, lanes>=16 hold K {8..15,24..31}
    a.q[0] = *(const v4u*)(arow + k0 + half * 8);
    a.q[1] = *(const v4u*)(arow + k0 + half * 8 + 16);
    // B 32x16 bf16 layout: lanes<16 hold K {0..7,16..23} of col lm, lanes>=16 hold K {8..15,24..31}
    b.q[0] = *(const v4u*)(brow + k0 + half * 8);
    b.q[1] = *(const v4u*)(brow + k0 + half * 8 + 16);
    if (k0 + 32 < Kd) {                      // WGP-scope prefetch (global_prefetch_b8)
      __builtin_prefetch(arow + k0 + 32, 0, 3);
      __builtin_prefetch(brow + k0 + 32, 0, 3);
    }
    acc = __builtin_amdgcn_wmma_f32_16x16x32_bf16(false, a.v, false, b.v,
                                                  (short)0, acc, false, false);
  }

  // C/D layout: VGPR r -> row (r + half*8), col lm
  float* crow = C + (long long)(tm + half * 8) * N + tn + lm;
#pragma unroll
  for (int r = 0; r < 8; ++r)
    crow[(long long)r * N] = acc[r];
}

// ---------- fused score + mask + softmax, emits bf16 attention weights ----------
// One block per (b, 8 q-rows): k-rows of kproj are streamed ONCE per block and
// reused against 8 q-rows cached in LDS (8x cut of the dominant L2 stream).
// Softmax: one q-row per wave, wave32 shuffle reductions.
__global__ void __launch_bounds__(256)
scores_softmax(const float* __restrict__ qp,   // [B*Q, H]
               const float* __restrict__ kp,   // [B*K, H]
               const float* __restrict__ Wv,   // [H]
               const int*   __restrict__ vlen, // [B]
               __bf16* __restrict__ attn) {    // [B*Q, K] bf16
  __shared__ float qs[QTILE * HH];   // 4 KB: 8 q-rows
  __shared__ float wv[HH];           // 0.5 KB
  __shared__ float sc[QTILE * KK];   // 16 KB: scores for 8 q-rows x 512 keys

  const int t  = threadIdx.x;
  const int b  = blockIdx.x >> 5;            // 32 q-tiles per batch
  const int q0 = (blockIdx.x & 31) * QTILE;

  // stage q-rows + Wv into LDS (vectorized)
  {
    const float4* src = (const float4*)(qp + (long long)((b << 8) + q0) * HH);
    ((float4*)qs)[t] = src[t];               // 256 * 4 floats = 8*128
    if (t < HH / 4) ((float4*)wv)[t] = ((const float4*)Wv)[t];
  }
  __syncthreads();

  const int vl = vlen[b];

  // each thread owns 2 key positions; stream k-row once, reuse across 8 q-rows
  for (int i = 0; i < 2; ++i) {
    const int k = t + (i << 8);
    const float4* krow4 = (const float4*)(kp + ((long long)(b << 9) + k) * HH);
    float acc[QTILE] = {};
    for (int h4 = 0; h4 < HH / 4; ++h4) {
      float4 kv = krow4[h4];
#pragma unroll
      for (int c = 0; c < 4; ++c) {
        const int h = h4 * 4 + c;
        const float kh = (&kv.x)[c];
        const float w  = wv[h];
#pragma unroll
        for (int qi = 0; qi < QTILE; ++qi)
          acc[qi] += w * fast_tanh(qs[qi * HH + h] + kh);
      }
    }
#pragma unroll
    for (int qi = 0; qi < QTILE; ++qi)
      sc[qi * KK + k] = (k < vl) ? acc[qi] : -1.0e6f;  // d2l masked_softmax fill
  }
  __syncthreads();

  // softmax: wave w handles q-row w; lane l owns 16 of the 512 scores
  const int w = t >> 5, l = t & 31;
  const float* row = sc + w * KK;
  float e[16];
  float m = -3.0e38f;
#pragma unroll
  for (int j = 0; j < 16; ++j) { e[j] = row[l + 32 * j]; m = fmaxf(m, e[j]); }
  for (int off = 16; off > 0; off >>= 1) m = fmaxf(m, __shfl_xor(m, off, 32));
  float ssum = 0.f;
#pragma unroll
  for (int j = 0; j < 16; ++j) { e[j] = __expf(e[j] - m); ssum += e[j]; }
  for (int off = 16; off > 0; off >>= 1) ssum += __shfl_xor(ssum, off, 32);
  const float inv = 1.0f / ssum;

  __bf16* arow = attn + (long long)((b << 8) + q0 + w) * KK;
#pragma unroll
  for (int j = 0; j < 16; ++j) arow[l + 32 * j] = f2bf(e[j] * inv);
}

extern "C" void kernel_launch(void* const* d_in, const int* in_sizes, int n_in,
                              void* d_out, int out_size, void* d_ws, size_t ws_size,
                              hipStream_t stream) {
  (void)in_sizes; (void)n_in; (void)out_size; (void)ws_size;
  const float* queries = (const float*)d_in[0];  // [8,256,256]
  const float* key     = (const float*)d_in[1];  // [8,512,256]
  const float* value   = (const float*)d_in[2];  // [8,512,256]
  const float* W_k     = (const float*)d_in[3];  // [128,256]
  const float* W_q     = (const float*)d_in[4];  // [128,256]
  const float* W_v     = (const float*)d_in[5];  // [1,128]
  const int*   vlen    = (const int*)d_in[6];    // [8]

  // workspace carve-up (256B aligned)
  char* ws = (char*)d_ws;
  size_t off = 0;
  auto carve = [&](size_t bytes) {
    void* p = ws + off;
    off = (off + bytes + 255) & ~(size_t)255;
    return p;
  };
  const int nQ  = BB * QQ * DD;   // 524288
  const int nK  = BB * KK * DD;   // 1048576
  const int nW  = HH * DD;        // 32768
  const int nV  = BB * KK * VV;   // 1048576
  const int nQP = BB * QQ * HH;   // 262144
  const int nKP = BB * KK * HH;   // 524288
  const int nAT = BB * QQ * KK;   // 1048576

  __bf16* qbf    = (__bf16*)carve((size_t)nQ  * 2);
  __bf16* kbf    = (__bf16*)carve((size_t)nK  * 2);
  __bf16* wqbf   = (__bf16*)carve((size_t)nW  * 2);
  __bf16* wkbf   = (__bf16*)carve((size_t)nW  * 2);
  __bf16* vtbf   = (__bf16*)carve((size_t)nV  * 2);
  float*  qproj  = (float*) carve((size_t)nQP * 4);
  float*  kproj  = (float*) carve((size_t)nKP * 4);
  __bf16* attnbf = (__bf16*)carve((size_t)nAT * 2);

  // 1) convert operands to bf16 (value also transposed to [B,V,K])
  cvt_f32_bf16<<<(nQ + 255) / 256, 256, 0, stream>>>(queries, qbf, nQ);
  cvt_f32_bf16<<<(nK + 255) / 256, 256, 0, stream>>>(key,     kbf, nK);
  cvt_f32_bf16<<<(nW + 255) / 256, 256, 0, stream>>>(W_q,     wqbf, nW);
  cvt_f32_bf16<<<(nW + 255) / 256, 256, 0, stream>>>(W_k,     wkbf, nW);
  transpose_cvt_value<<<(nV + 255) / 256, 256, 0, stream>>>(value, vtbf);

  // 2) projections: qproj[2048,128] = qbf[2048,256] * Wq^T ; kproj[4096,128] = kbf * Wk^T
  gemm_bf16_wmma<<<dim3(128, 1), 256, 0, stream>>>(qbf, 0, wqbf, 0, qproj, 0,
                                                   BB * QQ, HH, DD);
  gemm_bf16_wmma<<<dim3(256, 1), 256, 0, stream>>>(kbf, 0, wkbf, 0, kproj, 0,
                                                   BB * KK, HH, DD);

  // 3) fused tanh-score + mask + softmax -> bf16 attention [B*Q, 512]
  scores_softmax<<<BB * QQ / QTILE, 256, 0, stream>>>(qproj, kproj, W_v, vlen, attnbf);

  // 4) out[b] = attn[b] (256x512) * value[b] (512x256), batched over 8
  gemm_bf16_wmma<<<dim3(32, BB), 256, 0, stream>>>(
      attnbf, (long long)QQ * KK,
      vtbf,   (long long)VV * KK,
      (float*)d_out, (long long)QQ * VV,
      QQ, VV, KK);
}
